// TemporalGNN_27839978012783
// MI455X (gfx1250) — matrix operand order repaired and checked by reading, not compiled, not run on hardware
//
#include <hip/hip_runtime.h>
#include <hip/hip_bf16.h>

#define HID   32
#define HEADS 8
#define FEAT  (HID * HEADS)   // 256

typedef __attribute__((ext_vector_type(16))) __bf16 v16bf;
typedef __attribute__((ext_vector_type(8)))  float  v8f;

// ---------------------------------------------------------------------------
// WMMA GEMM: C[M,NC] = A[M,K] @ B   where B[k,n] is at B[k*BSK + n*BSN].
// One wave owns one 16-wide column stripe and MT consecutive 16-row tiles,
// holding all K/32 B fragments in registers across the M sweep.
// A fragment: 16-bit 16x32, contiguous K = 16*half + s (lane&15 = M row; half-wave = K group)
// B fragment: 32x16 layout (lane&15 = N col; K = 2*i + 16*half)
// D fragment: 8 VGPRs, M = r + 8*half, N = lane&15
// Requires M % 16 == 0 (true here: 50000 = 3125*16).
// ---------------------------------------------------------------------------
template<int K, int BSK, int BSN, int NC, int MT>
__global__ __launch_bounds__(32)
void gemm_wmma_bf16(const float* __restrict__ A, const float* __restrict__ B,
                    float* __restrict__ C, int M)
{
  constexpr int KB = K / 32;
  const int lane = threadIdx.x;        // 0..31
  const int l15  = lane & 15;
  const int half = lane >> 4;
  const int col  = blockIdx.y * 16 + l15;

  // load all B fragments once (reused for MT row tiles)
  v16bf bfrag[KB];
#pragma unroll
  for (int kb = 0; kb < KB; ++kb) {
#pragma unroll
    for (int i = 0; i < 8; ++i) {
      const int kk = kb * 32 + 2 * i + 16 * half;
      bfrag[kb][2*i]   = (__bf16)B[(size_t)kk       * BSK + (size_t)col * BSN];
      bfrag[kb][2*i+1] = (__bf16)B[(size_t)(kk + 1) * BSK + (size_t)col * BSN];
    }
  }

#pragma unroll
  for (int mt = 0; mt < MT; ++mt) {
    const int row0 = (blockIdx.x * MT + mt) * 16;
    if (row0 < M) {                       // uniform guard: EXEC stays all-ones
      const float* Ar = A + (size_t)(row0 + l15) * K;
      v8f acc = {};
#pragma unroll
      for (int kb = 0; kb < KB; ++kb) {
        v16bf a;
#pragma unroll
        for (int i = 0; i < 8; ++i) {
          const int ka = kb * 32 + 2*i + 16*half;
          a[2*i]   = (__bf16)Ar[ka];
          a[2*i+1] = (__bf16)Ar[ka + 1];
        }
        acc = __builtin_amdgcn_wmma_f32_16x16x32_bf16(false, a, false, bfrag[kb],
                                                      (short)0, acc, false, false);
      }
      float* Cr = C + (size_t)(row0 + 8 * half) * NC + col;
#pragma unroll
      for (int r = 0; r < 8; ++r)
        Cr[(size_t)r * NC] = acc[r];
    }
  }
}

// ---------------------------------------------------------------------------
// helpers
// ---------------------------------------------------------------------------
__device__ __forceinline__ void edge_nodes(int e, const int* __restrict__ ei,
                                           int E, int& s, int& d) {
  if (e < E) { s = ei[e]; d = ei[E + e]; } else { s = d = e - E; }
}
__device__ __forceinline__ unsigned mono(float f) {
  unsigned u = __float_as_uint(f);
  return (u & 0x80000000u) ? ~u : (u | 0x80000000u);
}
__device__ __forceinline__ float unmono(unsigned u) {
  u = (u & 0x80000000u) ? (u ^ 0x80000000u) : ~u;
  return __uint_as_float(u);
}
__device__ __forceinline__ float lrelu02(float v) { return v > 0.f ? v : 0.2f * v; }

// ---------------------------------------------------------------------------
// per-layer init: zero agg [N*256], emax=-inf (monotone code), denom=0, BN stats
// ---------------------------------------------------------------------------
__global__ void init_layer(float* agg, unsigned* emax, float* denom,
                           float* bnst, int N)
{
  const int i = blockIdx.x * 256 + threadIdx.x;
  if (i < N * FEAT)  agg[i] = 0.f;
  if (i < N * HEADS) { emax[i] = 0x007FFFFFu; denom[i] = 0.f; }
  if (i < 128)       bnst[i] = 0.f;
}

// alpha_s[n,h] = <xh[n,h,:], asrc[h,:]>, alpha_d likewise
__global__ void alpha_kernel(const float* __restrict__ xh,
                             const float* __restrict__ asrc,
                             const float* __restrict__ adst,
                             float* __restrict__ als, float* __restrict__ ald, int N)
{
  const int idx = blockIdx.x * 256 + threadIdx.x;
  if (idx >= N * HEADS) return;
  const int h = idx & (HEADS - 1);
  const int n = idx >> 3;
  const float* xr = xh + (size_t)n * FEAT + h * HID;
  const float* as = asrc + h * HID;
  const float* ad = adst + h * HID;
  float s = 0.f, d = 0.f;
#pragma unroll
  for (int c = 0; c < HID; ++c) { const float v = xr[c]; s += v * as[c]; d += v * ad[c]; }
  als[idx] = s; ald[idx] = d;
}

__global__ void edge_max(const int* __restrict__ ei, int E, int Etot,
                         const float* __restrict__ als, const float* __restrict__ ald,
                         unsigned* __restrict__ emax)
{
  const int idx = blockIdx.x * 256 + threadIdx.x;
  if (idx >= Etot * HEADS) return;
  const int h = idx & (HEADS - 1);
  const int e = idx >> 3;
  int s, d; edge_nodes(e, ei, E, s, d);
  const float v = lrelu02(als[s * HEADS + h] + ald[d * HEADS + h]);
  atomicMax(&emax[d * HEADS + h], mono(v));
}

__global__ void edge_sum(const int* __restrict__ ei, int E, int Etot,
                         const float* __restrict__ als, const float* __restrict__ ald,
                         const unsigned* __restrict__ emax, float* __restrict__ denom)
{
  const int idx = blockIdx.x * 256 + threadIdx.x;
  if (idx >= Etot * HEADS) return;
  const int h = idx & (HEADS - 1);
  const int e = idx >> 3;
  int s, d; edge_nodes(e, ei, E, s, d);
  const float v = lrelu02(als[s * HEADS + h] + ald[d * HEADS + h]);
  atomicAdd(&denom[d * HEADS + h], __expf(v - unmono(emax[d * HEADS + h])));
}

// agg[dst,:] += alpha(e,h) * xh[src,:] ; 64 threads/edge, 4 channels/thread
__global__ void edge_scatter(const int* __restrict__ ei, int E, int Etot,
                             const float* __restrict__ als, const float* __restrict__ ald,
                             const unsigned* __restrict__ emax, const float* __restrict__ denom,
                             const float* __restrict__ xh, float* __restrict__ agg)
{
  const int t = blockIdx.x * 256 + threadIdx.x;
  const int e = t >> 6;
  if (e >= Etot) return;
  const int c4 = (t & 63) * 4;          // 0..252
  const int h  = c4 >> 5;               // head
  int s, d; edge_nodes(e, ei, E, s, d);
  const float* xr = xh + (size_t)s * FEAT + c4;
  __builtin_prefetch(xr, 0, 0);         // global_prefetch_b8
  const float v  = lrelu02(als[s * HEADS + h] + ald[d * HEADS + h]);
  const float al = __expf(v - unmono(emax[d * HEADS + h])) / denom[d * HEADS + h];
  const float4 x4 = *(const float4*)xr;
  float* o = agg + (size_t)d * FEAT + c4;
  atomicAdd(o + 0, x4.x * al);
  atomicAdd(o + 1, x4.y * al);
  atomicAdd(o + 2, x4.z * al);
  atomicAdd(o + 3, x4.w * al);
}

// hpre[n,c] = mean_h agg[n,h,c] + bias[c]; LDS-reduced BN stats per channel
__global__ void mean_bias_stats(const float* __restrict__ agg,
                                const float* __restrict__ bias,
                                float* __restrict__ hpre,
                                float* __restrict__ bnsum, float* __restrict__ bnsqs, int N)
{
  __shared__ float ssum[256], ssq[256];
  const int idx = blockIdx.x * 256 + threadIdx.x;
  const int c = threadIdx.x & (HID - 1);
  float v = 0.f;
  if (idx < N * HID) {
    const int n = idx >> 5;
    const float* r = agg + (size_t)n * FEAT;
    float acc = 0.f;
#pragma unroll
    for (int h = 0; h < HEADS; ++h) acc += r[h * HID + c];
    v = acc * 0.125f + bias[c];
    hpre[idx] = v;
  }
  ssum[threadIdx.x] = v; ssq[threadIdx.x] = v * v;
  __syncthreads();
  for (int off = 128; off >= 32; off >>= 1) {
    if (threadIdx.x < off) {
      ssum[threadIdx.x] += ssum[threadIdx.x + off];
      ssq[threadIdx.x]  += ssq[threadIdx.x + off];
    }
    __syncthreads();
  }
  if (threadIdx.x < 32) {
    atomicAdd(&bnsum[c], ssum[threadIdx.x]);
    atomicAdd(&bnsqs[c], ssq[threadIdx.x]);
  }
}

__global__ void bn_finalize(const float* bnsum, const float* bnsqs,
                            const float* g, const float* b,
                            float* scale, float* shift, int N)
{
  const int c = threadIdx.x;
  if (c >= HID) return;
  const float invN = 1.f / (float)N;
  const float mu  = bnsum[c] * invN;
  const float var = bnsqs[c] * invN - mu * mu;
  const float sc  = g[c] * rsqrtf(var + 1e-5f);
  scale[c] = sc; shift[c] = b[c] - mu * sc;
}

// out = relu(bn(hpre)) (+ addsrc if given, for the residual h = h1 + h2)
__global__ void bn_relu(const float* __restrict__ hpre,
                        const float* __restrict__ scale, const float* __restrict__ shift,
                        float* __restrict__ out, const float* __restrict__ addsrc, int total)
{
  const int idx = blockIdx.x * 256 + threadIdx.x;
  if (idx >= total) return;
  const int c = idx & (HID - 1);
  float v = hpre[idx] * scale[c] + shift[c];
  v = v > 0.f ? v : 0.f;
  out[idx] = addsrc ? (v + addsrc[idx]) : v;
}

// GRUCell with h_prev = 0  ->  gh = bhh; hidden = (1-z) * tanh(i_n + r*bhh_n)
__global__ void gru_hidden(const float* __restrict__ gi,
                           const float* __restrict__ bih, const float* __restrict__ bhh,
                           float* __restrict__ hidden, int N)
{
  const int idx = blockIdx.x * 256 + threadIdx.x;
  if (idx >= N * HID) return;
  const int c = idx & (HID - 1);
  const int n = idx >> 5;
  const float* g = gi + (size_t)n * (3 * HID);
  const float ir = g[c]            + bih[c]            + bhh[c];
  const float iz = g[HID + c]      + bih[HID + c]      + bhh[HID + c];
  const float in_= g[2 * HID + c]  + bih[2 * HID + c];
  const float hn = bhh[2 * HID + c];
  const float r  = 1.f / (1.f + __expf(-ir));
  const float z  = 1.f / (1.f + __expf(-iz));
  const float nn = tanhf(in_ + r * hn);
  hidden[idx] = (1.f - z) * nn;
}

__global__ void out_linear(const float* __restrict__ hidden,
                           const float* __restrict__ ow, const float* __restrict__ ob,
                           float* __restrict__ out, int N)
{
  const int n = blockIdx.x * 256 + threadIdx.x;
  if (n >= N) return;
  const float* hr = hidden + (size_t)n * HID;
  float acc = ob[0];
#pragma unroll
  for (int c = 0; c < HID; ++c) acc += hr[c] * ow[c];
  out[n] = acc;
}

// ---------------------------------------------------------------------------
extern "C" void kernel_launch(void* const* d_in, const int* in_sizes, int n_in,
                              void* d_out, int out_size, void* d_ws, size_t ws_size,
                              hipStream_t stream)
{
  const float* x     = (const float*)d_in[0];
  const int*   ei    = (const int*)  d_in[1];
  const float* w1    = (const float*)d_in[2];
  const float* asrc1 = (const float*)d_in[3];
  const float* adst1 = (const float*)d_in[4];
  const float* b1    = (const float*)d_in[5];
  const float* bn1g  = (const float*)d_in[6];
  const float* bn1b  = (const float*)d_in[7];
  const float* w2    = (const float*)d_in[8];
  const float* asrc2 = (const float*)d_in[9];
  const float* adst2 = (const float*)d_in[10];
  const float* b2    = (const float*)d_in[11];
  const float* bn2g  = (const float*)d_in[12];
  const float* bn2b  = (const float*)d_in[13];
  const float* wih   = (const float*)d_in[14];
  /* d_in[15] = gru_whh: unused since h_prev == 0 */
  const float* bih   = (const float*)d_in[16];
  const float* bhh   = (const float*)d_in[17];
  const float* outw  = (const float*)d_in[18];
  const float* outb  = (const float*)d_in[19];

  const int N    = in_sizes[0] / 64;
  const int E    = in_sizes[1] / 2;
  const int Etot = E + N;

  // workspace carve-out
  char* ws = (char*)d_ws;
  size_t off = 0;
  auto carve = [&](size_t bytes) -> char* {
    char* p = ws + off; off += (bytes + 255) & ~(size_t)255; return p;
  };
  float*    xh    = (float*)   carve((size_t)N * FEAT * 4);
  float*    agg   = (float*)   carve((size_t)N * FEAT * 4);
  float*    alps  = (float*)   carve((size_t)N * HEADS * 4);
  float*    alpd  = (float*)   carve((size_t)N * HEADS * 4);
  unsigned* emax  = (unsigned*)carve((size_t)N * HEADS * 4);
  float*    denom = (float*)   carve((size_t)N * HEADS * 4);
  float*    hpre  = (float*)   carve((size_t)N * HID * 4);
  float*    h1    = (float*)   carve((size_t)N * HID * 4);
  float*    h     = (float*)   carve((size_t)N * HID * 4);
  float*    bnst  = (float*)   carve(128 * 4);
  float* bnsum = bnst, *bnsqs = bnst + 32, *bnsc = bnst + 64, *bnsh = bnst + 96;
  float* gi = xh;   // GRU GEMM output reuses xh region (N*96 <= N*256)

  float* out_lin = (float*)d_out;        // [N, 1]
  float* hidden  = (float*)d_out + N;    // [N, 32]

  const dim3 b256(256);
  const int gInit = (N * FEAT + 255) / 256;
  const int gEH   = (Etot * HEADS + 255) / 256;
  const int gScat = (Etot + 3) / 4;
  const int gNC   = (N * HID + 255) / 256;
  const int mT    = (N + 15) / 16;       // 16-row tiles (M % 16 == 0)
  const int mB    = (mT + 7) / 8;        // blocks of 8 M-tiles per wave

  for (int layer = 0; layer < 2; ++layer) {
    const float* as = layer ? asrc2 : asrc1;
    const float* ad = layer ? adst2 : adst1;
    const float* bb = layer ? b2 : b1;
    const float* gg = layer ? bn2g : bn1g;
    const float* gb = layer ? bn2b : bn1b;

    init_layer<<<gInit, b256, 0, stream>>>(agg, emax, denom, bnst, N);
    if (layer == 0)
      gemm_wmma_bf16<64, FEAT, 1, FEAT, 8>
        <<<dim3(mB, FEAT / 16), 32, 0, stream>>>(x, w1, xh, N);
    else
      gemm_wmma_bf16<32, FEAT, 1, FEAT, 8>
        <<<dim3(mB, FEAT / 16), 32, 0, stream>>>(h1, w2, xh, N);
    alpha_kernel <<<(N * HEADS + 255) / 256, b256, 0, stream>>>(xh, as, ad, alps, alpd, N);
    edge_max     <<<gEH,   b256, 0, stream>>>(ei, E, Etot, alps, alpd, emax);
    edge_sum     <<<gEH,   b256, 0, stream>>>(ei, E, Etot, alps, alpd, emax, denom);
    edge_scatter <<<gScat, b256, 0, stream>>>(ei, E, Etot, alps, alpd, emax, denom, xh, agg);
    mean_bias_stats<<<gNC, b256, 0, stream>>>(agg, bb, hpre, bnsum, bnsqs, N);
    bn_finalize  <<<1, 32, 0, stream>>>(bnsum, bnsqs, gg, gb, bnsc, bnsh, N);
    bn_relu      <<<gNC, b256, 0, stream>>>(hpre, bnsc, bnsh,
                                            layer ? h : h1,
                                            layer ? h1 : (const float*)nullptr, N * HID);
  }

  // GRU input GEMM: gi[N,96] = h[N,32] @ wih^T  (wih is [96,32] row-major)
  gemm_wmma_bf16<32, 1, HID, 3 * HID, 8>
    <<<dim3(mB, 96 / 16), 32, 0, stream>>>(h, wih, gi, N);
  gru_hidden <<<gNC, b256, 0, stream>>>(gi, bih, bhh, hidden, N);
  out_linear <<<(N + 255) / 256, b256, 0, stream>>>(hidden, outw, outb, out_lin, N);
}
